// LDPCBeliefPropagation_7146825580805
// MI455X (gfx1250) — matrix operand order repaired and checked by reading, not compiled, not run on hardware
//
#include <hip/hip_runtime.h>

typedef _Float16 v16h __attribute__((ext_vector_type(16)));
typedef _Float16 v8h  __attribute__((ext_vector_type(8)));
typedef float    v8f  __attribute__((ext_vector_type(8)));

#define B_    16
#define M_    512
#define N_    1024
#define MAXW  32
#define MAX_ITERS 5

// phi(x) = log(|tanh(x/2)|), computed as log((1-e^-|x|)/(1+e^-|x|)).
// x==0 -> log(0) = -inf, matching the reference's behavior on-support.
__device__ __forceinline__ float phi_f(float x) {
    float ax = fabsf(x);
    float e  = __expf(-ax);
    float t  = (1.0f - e) / (1.0f + e);
    return __logf(t);
}

__global__ void k_init_flags(int* flags) {
    flags[0] = 0;   // done
    flags[1] = 0;   // not_ok (syndrome violation seen this iteration)
}

// Build CSR of H: one thread per check row scans all N columns (ascending order).
__global__ void k_build_csr(const float* __restrict__ H,
                            int* __restrict__ row_cnt,
                            int* __restrict__ row_cols) {
    int m = blockIdx.x * blockDim.x + threadIdx.x;
    if (m >= M_) return;
    const float* row = H + (size_t)m * N_;
    int cnt = 0;
    for (int n = 0; n < N_; ++n) {
        if (row[n] != 0.0f) {
            if (cnt < MAXW) row_cols[m * MAXW + cnt] = n;
            ++cnt;
        }
    }
    row_cnt[m] = cnt > MAXW ? MAXW : cnt;
}

__global__ void k_h_to_f16(const float* __restrict__ H, _Float16* __restrict__ h16) {
    int i = blockIdx.x * blockDim.x + threadIdx.x;
    if (i < M_ * N_) h16[i] = (_Float16)H[i];
}

__global__ void k_init_llrtot(const float* __restrict__ llr,
                              float* __restrict__ llr_tot,
                              const int* __restrict__ flags) {
    if (flags[0]) return;                      // frozen: keep previous llr_tot
    int i = blockIdx.x * blockDim.x + threadIdx.x;
    if (i < B_ * N_) llr_tot[i] = llr[i];
}

// Check-node update: per (b, m) compute extrinsic phi-sum and sign product over
// the row's neighbors, emit c2v messages, accumulate into llr_tot atomically.
// Two passes (recompute phi) instead of a local array -> no scratch spills.
__global__ void k_check_node(const float* __restrict__ llr,
                             const float* __restrict__ v2c,
                             float* __restrict__ c2v,
                             float* __restrict__ llr_tot,
                             const int* __restrict__ row_cnt,
                             const int* __restrict__ row_cols,
                             const int* __restrict__ flags,
                             int first_iter) {
    if (flags[0]) return;
    int tid = blockIdx.x * blockDim.x + threadIdx.x;
    if (tid >= B_ * M_) return;
    int b = tid & (B_ - 1);
    int m = tid >> 4;
    int w = row_cnt[m];

    float s = 0.0f;
    float sprod = 1.0f;
    for (int i = 0; i < w; ++i) {
        int c = row_cols[m * MAXW + i];
        float x = first_iter ? llr[b * N_ + c] : v2c[(m * MAXW + i) * B_ + b];
        s += phi_f(x);
        if (x < 0.0f) sprod = -sprod;
    }
    for (int i = 0; i < w; ++i) {
        int c = row_cols[m * MAXW + i];
        float x = first_iter ? llr[b * N_ + c] : v2c[(m * MAXW + i) * B_ + b];
        float p  = phi_f(x);
        float sg = (x < 0.0f) ? -1.0f : 1.0f;
        float msg = -phi_f(s - p) * (sprod * sg);   // extrinsic c2v message
        c2v[(m * MAXW + i) * B_ + b] = msg;
        // Native GLOBAL_ATOMIC_ADD_F32 (no CAS loop): llr_tot is coarse-grained
        // device memory; fp32 sum order already differs from JAX's reduction.
        unsafeAtomicAdd(&llr_tot[b * N_ + c], msg);
    }
}

// Variable-node update on the support: v2c = llr_tot - c2v (== H*llr_tot - c2v).
__global__ void k_var_node(const float* __restrict__ llr_tot,
                           const float* __restrict__ c2v,
                           float* __restrict__ v2c,
                           const int* __restrict__ row_cnt,
                           const int* __restrict__ row_cols,
                           const int* __restrict__ flags) {
    if (flags[0]) return;
    int tid = blockIdx.x * blockDim.x + threadIdx.x;
    if (tid >= B_ * M_) return;
    int b = tid & (B_ - 1);
    int m = tid >> 4;
    int w = row_cnt[m];
    for (int i = 0; i < w; ++i) {
        int c = row_cols[m * MAXW + i];
        int e = (m * MAXW + i) * B_ + b;
        v2c[e] = llr_tot[b * N_ + c] - c2v[e];
    }
}

__global__ void k_est_f16(const float* __restrict__ llr_tot,
                          _Float16* __restrict__ est,
                          const int* __restrict__ flags) {
    if (flags[0]) return;
    int i = blockIdx.x * blockDim.x + threadIdx.x;
    if (i < B_ * N_) est[i] = (llr_tot[i] < 0.0f) ? (_Float16)1.0f : (_Float16)0.0f;
}

// Syndrome S[b, m0+j] = sum_n est[b,n] * H[m0+j, n] via v_wmma_f32_16x16x32_f16.
// One wave (32 lanes) per 16-check tile; K-loop over N in chunks of 32.
// A (est, 16x32 f16): lanes 0-15 are M=lane with K in {0..7}U{16..23};
//                     lanes 16-31 are M=lane-16 with K in {8..15}U{24..31}.
// B (H^T slice, 32x16 f16): lane holds check column m0+(lane&15); same K split
//                     as A: lanes 0-15 K {0..7}U{16..23}, lanes 16-31 K {8..15}U{24..31}.
__global__ void __launch_bounds__(32)
k_syndrome_wmma(const _Float16* __restrict__ est,
                const _Float16* __restrict__ h16,
                int* __restrict__ flags) {
    if (flags[0]) return;
    int lane   = threadIdx.x;
    int m0     = blockIdx.x * 16;
    int row    = lane & 15;
    int hiHalf = lane >> 4;

    const _Float16* arow = est + row * N_;                   // batch row
    const _Float16* brow = h16 + (size_t)(m0 + row) * N_;    // check row

    v8f acc = {};
    for (int k0 = 0; k0 < N_; k0 += 32) {
        int kA = k0 + hiHalf * 8;
        v8h lo = *(const v8h*)(arow + kA);
        v8h hi = *(const v8h*)(arow + kA + 16);
        v16h a;
#pragma unroll
        for (int i = 0; i < 8; ++i) { a[i] = lo[i]; a[i + 8] = hi[i]; }

        int kB = k0 + hiHalf * 8;
        v8h blo = *(const v8h*)(brow + kB);
        v8h bhi = *(const v8h*)(brow + kB + 16);
        v16h bm;
#pragma unroll
        for (int i = 0; i < 8; ++i) { bm[i] = blo[i]; bm[i + 8] = bhi[i]; }

        acc = __builtin_amdgcn_wmma_f32_16x16x32_f16(
                  false, a, false, bm, (short)0, acc, false, false);
    }

    // Counts are exact small integers; parity odd => violated check.
    int viol = 0;
#pragma unroll
    for (int r = 0; r < 8; ++r) {
        int v = (int)(acc[r] + 0.5f);
        viol |= (v & 1);
    }
    if (viol) atomicOr(&flags[1], 1);
}

// done' = done | (no violations this iteration); reset violation flag.
__global__ void k_done_update(int* flags) {
    if (flags[1] == 0) flags[0] = 1;
    flags[1] = 0;
}

extern "C" void kernel_launch(void* const* d_in, const int* in_sizes, int n_in,
                              void* d_out, int out_size, void* d_ws, size_t ws_size,
                              hipStream_t stream) {
    const float* llr = (const float*)d_in[0];   // [B,1,N]
    const float* H   = (const float*)d_in[1];   // [1,M,N]
    float* llr_tot   = (float*)d_out;           // [B,1,N] output, used in-place

    char* p = (char*)d_ws;
    int*      row_cnt  = (int*)p;        p += M_ * sizeof(int);
    int*      row_cols = (int*)p;        p += (size_t)M_ * MAXW * sizeof(int);
    float*    c2v      = (float*)p;      p += (size_t)B_ * M_ * MAXW * sizeof(float);
    float*    v2c      = (float*)p;      p += (size_t)B_ * M_ * MAXW * sizeof(float);
    _Float16* h16      = (_Float16*)p;   p += (size_t)M_ * N_ * sizeof(_Float16);
    _Float16* est      = (_Float16*)p;   p += (size_t)B_ * N_ * sizeof(_Float16);
    int*      flags    = (int*)p;

    k_init_flags<<<1, 1, 0, stream>>>(flags);
    k_build_csr<<<(M_ + 255) / 256, 256, 0, stream>>>(H, row_cnt, row_cols);
    k_h_to_f16<<<(M_ * N_ + 255) / 256, 256, 0, stream>>>(H, h16);

    for (int it = 0; it < MAX_ITERS; ++it) {
        k_init_llrtot<<<(B_ * N_ + 255) / 256, 256, 0, stream>>>(llr, llr_tot, flags);
        k_check_node<<<(B_ * M_ + 255) / 256, 256, 0, stream>>>(
            llr, v2c, c2v, llr_tot, row_cnt, row_cols, flags, it == 0 ? 1 : 0);
        k_est_f16<<<(B_ * N_ + 255) / 256, 256, 0, stream>>>(llr_tot, est, flags);
        k_syndrome_wmma<<<M_ / 16, 32, 0, stream>>>(est, h16, flags);
        k_var_node<<<(B_ * M_ + 255) / 256, 256, 0, stream>>>(
            llr_tot, c2v, v2c, row_cnt, row_cols, flags);
        k_done_update<<<1, 1, 0, stream>>>(flags);
    }
}
